// SparseMoEBlock_40999757807880
// MI455X (gfx1250) — matrix operand order repaired and checked by reading, not compiled, not run on hardware
//
#include <hip/hip_runtime.h>
#include <hip/hip_bf16.h>
#include <math.h>

#define T   4096
#define H   1024
#define II  2048
#define E   8
#define MT  64          // tokens per workgroup tile
#define KT  32          // K step (matches wmma bf16 K)
#define NT  128         // N columns per workgroup tile
#define CAPMT (T / MT)  // worst-case M tiles per expert

typedef __attribute__((ext_vector_type(16))) __bf16 v16bf;
typedef __attribute__((ext_vector_type(8)))  __bf16 v8bf;
typedef __attribute__((ext_vector_type(2)))  __bf16 v2bf;
typedef __attribute__((ext_vector_type(8)))  float  v8f;

// ---------------------------------------------------------------- init
__global__ void init_kernel(int* counts) {
    if (threadIdx.x < E) counts[threadIdx.x] = 0;
}

// ---------------------------------------------------------------- router
// one block per token; 8 waves, wave e computes logit for expert e
__global__ __launch_bounds__(256)
void router_kernel(const float* __restrict__ x, const float* __restrict__ gate_w,
                   float* __restrict__ logits_out, int* __restrict__ topi,
                   float* __restrict__ topw, int* __restrict__ counts) {
    const int t    = blockIdx.x;
    const int wave = threadIdx.x >> 5;
    const int lane = threadIdx.x & 31;
    const float* xr = x + (size_t)t * H;
    const float* gr = gate_w + (size_t)wave * H;
    float s = 0.f;
    for (int h = lane; h < H; h += 32) s += xr[h] * gr[h];
    #pragma unroll
    for (int off = 16; off > 0; off >>= 1) s += __shfl_xor(s, off, 32);
    __shared__ float lg[E];
    if (lane == 0) lg[wave] = s;
    __syncthreads();
    if (threadIdx.x == 0) {
        float l[E];
        float mx = -INFINITY;
        #pragma unroll
        for (int e = 0; e < E; ++e) { l[e] = lg[e]; logits_out[(size_t)t * E + e] = l[e]; if (l[e] > mx) mx = l[e]; }
        // top-2 (first max wins ties, like lax.top_k)
        int i0 = 0; float l0 = l[0];
        #pragma unroll
        for (int e = 1; e < E; ++e) if (l[e] > l0) { l0 = l[e]; i0 = e; }
        int i1 = -1; float l1 = -INFINITY;
        #pragma unroll
        for (int e = 0; e < E; ++e) if (e != i0 && l[e] > l1) { l1 = l[e]; i1 = e; }
        float p0 = __expf(l0 - mx), p1 = __expf(l1 - mx);
        float inv = 1.f / (p0 + p1);
        topi[2 * t + 0] = i0;  topw[2 * t + 0] = p0 * inv;
        topi[2 * t + 1] = i1;  topw[2 * t + 1] = p1 * inv;
        atomicAdd(&counts[i0], 1);
        atomicAdd(&counts[i1], 1);
    }
}

// ---------------------------------------------------------------- offsets
__global__ void offsets_kernel(const int* __restrict__ counts, int* __restrict__ offs,
                               int* __restrict__ cursor) {
    if (threadIdx.x == 0) {
        int acc = 0;
        for (int e = 0; e < E; ++e) { offs[e] = acc; acc += counts[e]; }
    }
    if (threadIdx.x < E) cursor[threadIdx.x] = 0;
}

// ---------------------------------------------------------------- assign
__global__ void assign_kernel(const int* __restrict__ topi, int* __restrict__ cursor,
                              const int* __restrict__ offs, int* __restrict__ pairrow,
                              int* __restrict__ rowtok) {
    int t = blockIdx.x * blockDim.x + threadIdx.x;
    if (t >= T) return;
    #pragma unroll
    for (int k = 0; k < 2; ++k) {
        int e = topi[2 * t + k];
        int slot = atomicAdd(&cursor[e], 1);
        int r = offs[e] + slot;
        rowtok[r] = t;
        pairrow[2 * t + k] = r;
    }
}

// ---------------------------------------------------------------- gate+up (fused, SiLU epilogue)
// grid: (E*CAPMT, II/NT); block 256 (8 waves -> 4x2 grid of 16x64 wave tiles)
// double-buffered LDS, register-staged global prefetch
__global__ __launch_bounds__(256)
void gateup_kernel(const float* __restrict__ x, const float* __restrict__ w_gate,
                   const float* __restrict__ w_up, const int* __restrict__ counts,
                   const int* __restrict__ offs, const int* __restrict__ rowtok,
                   __bf16* __restrict__ hbuf) {
    const int e   = blockIdx.x / CAPMT;
    const int mt  = blockIdx.x % CAPMT;
    const int cnt = counts[e];
    const int m_base = mt * MT;
    if (m_base >= cnt) return;                 // block-uniform exit
    const int off   = offs[e];
    const int nbase = blockIdx.y * NT;

    __shared__ __bf16 xs [2][MT * KT];         // [m][k]
    __shared__ __bf16 wgT[2][NT * KT];         // [n][k] (transposed)
    __shared__ __bf16 wuT[2][NT * KT];

    const int tid  = threadIdx.x;
    const int wave = tid >> 5, lane = tid & 31;
    const int wm = wave & 3, wn = wave >> 2;
    const int lm = lane & 15, hig = lane >> 4;

    v8f accg[4] = {v8f{}, v8f{}, v8f{}, v8f{}};
    v8f accu[4] = {v8f{}, v8f{}, v8f{}, v8f{}};

    int toks[4];
    #pragma unroll
    for (int i = 0; i < 4; ++i) {
        int p = i * 256 + tid;                 // 0..1023 (pairs)
        int m = p >> 4;
        toks[i] = (m_base + m < cnt) ? rowtok[off + m_base + m] : rowtok[off];
    }
    const float* wg_base = w_gate + (size_t)e * H * II;
    const float* wu_base = w_up   + (size_t)e * H * II;

    float2 xr[4];
    float  wgr[16], wur[16];

    auto load_tile = [&](int kb) {
        #pragma unroll
        for (int i = 0; i < 4; ++i) {
            int p = i * 256 + tid;
            int k = (p & 15) * 2;
            xr[i] = *(const float2*)(x + (size_t)toks[i] * H + kb + k);
        }
        #pragma unroll
        for (int i = 0; i < 8; ++i) {
            int p = i * 256 + tid;             // 0..2047 (pairs)
            int n = p & 127;
            int k = (p >> 7) * 2;              // 0,2,..,30
            size_t g0 = (size_t)(kb + k) * II + nbase + n;
            wgr[2 * i]     = wg_base[g0];
            wgr[2 * i + 1] = wg_base[g0 + II];
            wur[2 * i]     = wu_base[g0];
            wur[2 * i + 1] = wu_base[g0 + II];
        }
    };
    auto store_tile = [&](int b) {
        #pragma unroll
        for (int i = 0; i < 4; ++i) {
            int p = i * 256 + tid;
            int m = p >> 4, k = (p & 15) * 2;
            v2bf v = { (__bf16)xr[i].x, (__bf16)xr[i].y };
            *(v2bf*)&xs[b][m * KT + k] = v;
        }
        #pragma unroll
        for (int i = 0; i < 8; ++i) {
            int p = i * 256 + tid;
            int n = p & 127, k = (p >> 7) * 2;
            v2bf vg = { (__bf16)wgr[2 * i], (__bf16)wgr[2 * i + 1] };
            v2bf vu = { (__bf16)wur[2 * i], (__bf16)wur[2 * i + 1] };
            *(v2bf*)&wgT[b][n * KT + k] = vg;
            *(v2bf*)&wuT[b][n * KT + k] = vu;
        }
    };

    load_tile(0);
    store_tile(0);
    int buf = 0;
    for (int kb = 0; kb < H; kb += KT) {
        __syncthreads();
        if (kb + KT < H) load_tile(kb + KT);   // prefetch next tile into registers

        // A fragment (contiguous K: lanes 0-15 K=0..15, lanes 16-31 K=16..31)
        const __bf16* xrow = &xs[buf][(wm * 16 + lm) * KT];
        v8bf a0 = *(const v8bf*)(xrow + hig * 16);
        v8bf a1 = *(const v8bf*)(xrow + hig * 16 + 8);
        v16bf a = __builtin_shufflevector(a0, a1, 0,1,2,3,4,5,6,7,8,9,10,11,12,13,14,15);

        #pragma unroll
        for (int ns = 0; ns < 4; ++ns) {
            int n = wn * 64 + ns * 16 + lm;
            v16bf bg = *(const v16bf*)&wgT[buf][n * KT + hig * 16];
            v16bf bu = *(const v16bf*)&wuT[buf][n * KT + hig * 16];
            accg[ns] = __builtin_amdgcn_wmma_f32_16x16x32_bf16(false, a, false, bg,
                                                               (short)0, accg[ns], false, false);
            accu[ns] = __builtin_amdgcn_wmma_f32_16x16x32_bf16(false, a, false, bu,
                                                               (short)0, accu[ns], false, false);
        }
        if (kb + KT < H) store_tile(buf ^ 1);  // fill other buffer while peers still read this one
        buf ^= 1;
    }

    #pragma unroll
    for (int ns = 0; ns < 4; ++ns) {
        int col = nbase + wn * 64 + ns * 16 + lm;
        #pragma unroll
        for (int r = 0; r < 8; ++r) {
            int ml = wm * 16 + r + hig * 8;
            if (m_base + ml < cnt) {
                float g = accg[ns][r], u = accu[ns][r];
                float sig = __builtin_amdgcn_rcpf(1.f + __expf(-g));   // fast sigmoid
                hbuf[(size_t)(off + m_base + ml) * II + col] = (__bf16)(g * sig * u);
            }
        }
    }
}

// ---------------------------------------------------------------- down proj
// grid: (E*CAPMT, H/NT)
__global__ __launch_bounds__(256)
void down_kernel(const __bf16* __restrict__ hbuf, const float* __restrict__ w_down,
                 const int* __restrict__ counts, const int* __restrict__ offs,
                 float* __restrict__ ybuf) {
    const int e   = blockIdx.x / CAPMT;
    const int mt  = blockIdx.x % CAPMT;
    const int cnt = counts[e];
    const int m_base = mt * MT;
    if (m_base >= cnt) return;
    const int off   = offs[e];
    const int nbase = blockIdx.y * NT;

    __shared__ __bf16 xs [2][MT * KT];
    __shared__ __bf16 wdT[2][NT * KT];

    const int tid  = threadIdx.x;
    const int wave = tid >> 5, lane = tid & 31;
    const int wm = wave & 3, wn = wave >> 2;
    const int lm = lane & 15, hig = lane >> 4;

    v8f acc[4] = {v8f{}, v8f{}, v8f{}, v8f{}};

    int rows[4];
    #pragma unroll
    for (int i = 0; i < 4; ++i) {
        int p = i * 256 + tid;
        int mm = m_base + (p >> 4);
        if (mm >= cnt) mm = cnt - 1;
        rows[i] = off + mm;
    }
    const float* wd_base = w_down + (size_t)e * II * H;

    unsigned int xr[4];                        // packed bf16 pairs (already bf16 in hbuf)
    float wdr[16];

    auto load_tile = [&](int kb) {
        #pragma unroll
        for (int i = 0; i < 4; ++i) {
            int p = i * 256 + tid;
            int k = (p & 15) * 2;
            xr[i] = *(const unsigned int*)(hbuf + (size_t)rows[i] * II + kb + k);
        }
        #pragma unroll
        for (int i = 0; i < 8; ++i) {
            int p = i * 256 + tid;
            int n = p & 127;
            int k = (p >> 7) * 2;
            size_t g0 = (size_t)(kb + k) * H + nbase + n;
            wdr[2 * i]     = wd_base[g0];
            wdr[2 * i + 1] = wd_base[g0 + H];
        }
    };
    auto store_tile = [&](int b) {
        #pragma unroll
        for (int i = 0; i < 4; ++i) {
            int p = i * 256 + tid;
            int m = p >> 4, k = (p & 15) * 2;
            *(unsigned int*)&xs[b][m * KT + k] = xr[i];
        }
        #pragma unroll
        for (int i = 0; i < 8; ++i) {
            int p = i * 256 + tid;
            int n = p & 127, k = (p >> 7) * 2;
            v2bf v = { (__bf16)wdr[2 * i], (__bf16)wdr[2 * i + 1] };
            *(v2bf*)&wdT[b][n * KT + k] = v;
        }
    };

    load_tile(0);
    store_tile(0);
    int buf = 0;
    for (int kb = 0; kb < II; kb += KT) {
        __syncthreads();
        if (kb + KT < II) load_tile(kb + KT);

        const __bf16* xrow = &xs[buf][(wm * 16 + lm) * KT];
        v8bf a0 = *(const v8bf*)(xrow + hig * 16);
        v8bf a1 = *(const v8bf*)(xrow + hig * 16 + 8);
        v16bf a = __builtin_shufflevector(a0, a1, 0,1,2,3,4,5,6,7,8,9,10,11,12,13,14,15);

        #pragma unroll
        for (int ns = 0; ns < 4; ++ns) {
            int n = wn * 64 + ns * 16 + lm;
            v16bf b = *(const v16bf*)&wdT[buf][n * KT + hig * 16];
            acc[ns] = __builtin_amdgcn_wmma_f32_16x16x32_bf16(false, a, false, b,
                                                              (short)0, acc[ns], false, false);
        }
        if (kb + KT < II) store_tile(buf ^ 1);
        buf ^= 1;
    }

    #pragma unroll
    for (int ns = 0; ns < 4; ++ns) {
        int col = nbase + wn * 64 + ns * 16 + lm;
        #pragma unroll
        for (int r = 0; r < 8; ++r) {
            int ml = wm * 16 + r + hig * 8;
            if (m_base + ml < cnt)
                ybuf[(size_t)(off + m_base + ml) * H + col] = acc[ns][r];
        }
    }
}

// ---------------------------------------------------------------- combine
__global__ __launch_bounds__(256)
void combine_kernel(const float* __restrict__ ybuf, const int* __restrict__ pairrow,
                    const float* __restrict__ topw, float* __restrict__ out) {
    const int t = blockIdx.x;
    const int r0 = pairrow[2 * t], r1 = pairrow[2 * t + 1];
    const float w0 = topw[2 * t], w1 = topw[2 * t + 1];
    const float* y0 = ybuf + (size_t)r0 * H;
    const float* y1 = ybuf + (size_t)r1 * H;
    float* o = out + (size_t)t * H;
    for (int h = threadIdx.x; h < H; h += blockDim.x)
        o[h] = w0 * y0[h] + w1 * y1[h];
}

// ---------------------------------------------------------------- launch
extern "C" void kernel_launch(void* const* d_in, const int* in_sizes, int n_in,
                              void* d_out, int out_size, void* d_ws, size_t ws_size,
                              hipStream_t stream) {
    const float* x      = (const float*)d_in[0];
    const float* gate_w = (const float*)d_in[1];
    const float* w_gate = (const float*)d_in[2];
    const float* w_up   = (const float*)d_in[3];
    const float* w_down = (const float*)d_in[4];

    float* outp   = (float*)d_out;
    float* logits = outp + (size_t)T * H;      // second tuple output

    // workspace carve
    int*   counts  = (int*)d_ws;               // [8]
    int*   offs    = counts + 8;               // [8]
    int*   cursor  = offs + 8;                 // [8]
    int*   topi    = cursor + 8;               // [2T]
    float* topw    = (float*)(topi + 2 * T);   // [2T]
    int*   pairrow = (int*)(topw + 2 * T);     // [2T]
    int*   rowtok  = pairrow + 2 * T;          // [2T]
    __bf16* hbuf   = (__bf16*)(rowtok + 2 * T);           // [2T][II] bf16 (~33.5 MB)
    float*  ybuf   = (float*)(hbuf + (size_t)2 * T * II); // [2T][H] f32  (~33.5 MB)

    init_kernel<<<1, 32, 0, stream>>>(counts);
    router_kernel<<<T, 256, 0, stream>>>(x, gate_w, logits, topi, topw, counts);
    offsets_kernel<<<1, 32, 0, stream>>>(counts, offs, cursor);
    assign_kernel<<<T / 256, 256, 0, stream>>>(topi, cursor, offs, pairrow, rowtok);
    gateup_kernel<<<dim3(E * CAPMT, II / NT), 256, 0, stream>>>(x, w_gate, w_up,
                                                                counts, offs, rowtok, hbuf);
    down_kernel<<<dim3(E * CAPMT, H / NT), 256, 0, stream>>>(hbuf, w_down, counts, offs, ybuf);
    combine_kernel<<<T, 256, 0, stream>>>(ybuf, pairrow, topw, outp);
}
